// GraphVAE_36876589203650
// MI455X (gfx1250) — compile-verified, matrix operand order checked by reading, not run on hardware
//
#include <hip/hip_runtime.h>
#include <hip/hip_bf16.h>
#include <math.h>

typedef _Float16 h16;
typedef __attribute__((ext_vector_type(16))) _Float16 v16h;
typedef __attribute__((ext_vector_type(8)))  float    v8f;

// ---------------- problem constants ----------------
constexpr int ND  = 50000;    // nodes
constexpr int EG  = 800000;   // edges
constexpr int NG  = 1024;     // graphs
constexpr int FEAT = 7;
constexpr int HD  = 128;      // HID
constexpr int LT  = 64;       // LAT
constexpr int MXN = 64;
constexpr float BNF = 0.99999500003749937f;       // 1/sqrt(1+1e-5)
constexpr float INV_SQRT_HID = 0.08838834764831843f;

// ---------------- workspace layout (float slots) ----------------
constexpr size_t NH    = (size_t)ND * HD;               // 6.4M
constexpr size_t O_H   = 0;
constexpr size_t O_T1  = O_H   + NH;                    // [ND,256]
constexpr size_t O_T2  = O_T1  + 2 * NH;                // [ND,128]
constexpr size_t O_T3  = O_T2  + NH;                    // [ND,128]
constexpr size_t O_LG  = O_T3  + NH;                    // (EG+ND)*4 edge logits
constexpr size_t O_M   = O_LG  + (size_t)(EG + ND) * 4; // [ND,4] seg max
constexpr size_t O_S   = O_M   + (size_t)ND * 4;        // [ND,4] seg sum
constexpr size_t O_DEG = O_S   + (size_t)ND * 4;        // [ND] dinv
constexpr size_t O_GM  = O_DEG + ND;                    // [NG,128]
constexpr size_t O_GV  = O_GM  + (size_t)NG * HD;
constexpr size_t O_GC  = O_GV  + (size_t)NG * HD;       // [NG]
constexpr size_t O_GH  = O_GC  + NG;                    // [NG,128] pooled
constexpr size_t O_U1  = O_GH  + (size_t)NG * HD;       // [NG,512]
constexpr size_t O_U2  = O_U1  + (size_t)NG * 512;      // [NG,512]
constexpr size_t O_NEB = O_U2  + (size_t)NG * 512;      // node_embeds [NG*64,128]
constexpr size_t O_F16A = O_NEB + (size_t)NG * MXN * HD; // 12.8M halfs (A staging)
constexpr size_t O_F16W = O_F16A + NH;                  // 4.19M halfs (W staging)

// ---------------- d_out layout ----------------
constexpr size_t OUT_NL = 0;                                  // 1024*64*7
constexpr size_t OUT_EL = OUT_NL + (size_t)NG * MXN * FEAT;   // 1024*64*64
constexpr size_t OUT_NC = OUT_EL + (size_t)NG * MXN * MXN;    // 1024*64
constexpr size_t OUT_MU = OUT_NC + (size_t)NG * MXN;          // 1024*64
constexpr size_t OUT_LV = OUT_MU + (size_t)NG * LT;           // 1024*64

// ================= device helpers =================
__device__ __forceinline__ void atomicMaxF(float* addr, float val) {
  unsigned int* ua = (unsigned int*)addr;
  unsigned int old = __float_as_uint(*addr);
  while (__uint_as_float(old) < val) {
    unsigned int assumed = old;
    old = atomicCAS(ua, assumed, __float_as_uint(val));
    if (old == assumed) break;
  }
}

// A/B fragment for v_wmma_f32_16x16x32_f16 (wave32).
// 16-bit A 16x32 layout: lane L (L&15 = row-in-tile), lane>>4 selects K-half.
// elems 0..7  <-> K = k0 + (lane>>4)*8 + 0..7
// elems 8..15 <-> K = k0 + 16 + (lane>>4)*8 + 0..7
// B = W^T with row-major W[N,K] uses identical addressing on rows of W.
__device__ __forceinline__ v16h frag_load_nc(const h16* __restrict__ base, int rowBase,
                                             int ld, int k0, int lane) {
  int r = rowBase + (lane & 15);
  int g = (lane >> 4) << 3;
  const h16* p = base + (size_t)r * ld + k0 + g;
  v16h f;
#pragma unroll
  for (int i = 0; i < 8; ++i) { f[i] = p[i]; f[i + 8] = p[16 + i]; }
  return f;
}

__device__ __forceinline__ v16h frag_load(const h16* __restrict__ base, int rowBase,
                                          int nRows, int ld, int k0, int lane) {
  int r = rowBase + (lane & 15);
  int g = (lane >> 4) << 3;
  v16h f;
  if (r < nRows) {
    const h16* p = base + (size_t)r * ld + k0 + g;
#pragma unroll
    for (int i = 0; i < 8; ++i) { f[i] = p[i]; f[i + 8] = p[16 + i]; }
  } else {
#pragma unroll
    for (int i = 0; i < 16; ++i) f[i] = (h16)0.f;
  }
  return f;
}

// ================= kernels =================
__global__ void k_f2h(const float* __restrict__ s, h16* __restrict__ d, int n) {
  int i = blockIdx.x * blockDim.x + threadIdx.x;
  if (i < n) d[i] = (h16)s[i];
}

__global__ void k_fill(float* p, float v, int n) {
  int i = blockIdx.x * blockDim.x + threadIdx.x;
  if (i < n) p[i] = v;
}

__global__ void k_lin_in(const float* __restrict__ x, const float* __restrict__ w,
                         const float* __restrict__ b, float* __restrict__ h) {
  int idx = blockIdx.x * blockDim.x + threadIdx.x;
  if (idx >= ND * HD) return;
  int i = idx >> 7, o = idx & 127;
  float acc = b[o];
#pragma unroll
  for (int k = 0; k < FEAT; ++k) acc += x[i * FEAT + k] * w[o * FEAT + k];
  h[idx] = acc;
}

// Fast path: C[M,N] = scale*act(A @ W^T + b); requires M%16==0, N%64==0.
// One wave computes a 16x64 tile: A fragment reused across 4 B tiles.
__global__ void k_gemm_t64(const h16* __restrict__ A, const h16* __restrict__ W,
                           const float* __restrict__ bias, float* __restrict__ C,
                           int M, int N, int K, int relu, float scale) {
  int lane = threadIdx.x;
  int nBase = blockIdx.x * 64, mBase = blockIdx.y * 16;
  v8f acc0 = {}, acc1 = {}, acc2 = {}, acc3 = {};
  for (int k0 = 0; k0 < K; k0 += 32) {
    v16h a  = frag_load_nc(A, mBase, K, k0, lane);
    v16h b0 = frag_load_nc(W, nBase +  0, K, k0, lane);
    v16h b1 = frag_load_nc(W, nBase + 16, K, k0, lane);
    v16h b2 = frag_load_nc(W, nBase + 32, K, k0, lane);
    v16h b3 = frag_load_nc(W, nBase + 48, K, k0, lane);
    acc0 = __builtin_amdgcn_wmma_f32_16x16x32_f16(false, a, false, b0, (short)0, acc0, false, false);
    acc1 = __builtin_amdgcn_wmma_f32_16x16x32_f16(false, a, false, b1, (short)0, acc1, false, false);
    acc2 = __builtin_amdgcn_wmma_f32_16x16x32_f16(false, a, false, b2, (short)0, acc2, false, false);
    acc3 = __builtin_amdgcn_wmma_f32_16x16x32_f16(false, a, false, b3, (short)0, acc3, false, false);
  }
  int row0 = mBase + ((lane >> 4) << 3);  // C layout: vgpr r -> M = r + 8*(lane>>4)
  v8f* accs[4] = {&acc0, &acc1, &acc2, &acc3};
#pragma unroll
  for (int s = 0; s < 4; ++s) {
    int col = nBase + s * 16 + (lane & 15);
    float bv = bias ? bias[col] : 0.f;
    v8f& ac = *accs[s];
#pragma unroll
    for (int r = 0; r < 8; ++r) {
      float v = ac[r] + bv;
      if (relu) v = v > 0.f ? v : 0.f;
      C[(size_t)(row0 + r) * N + col] = v * scale;
    }
  }
}

// Generic guarded path (used only for N=7 nt projection).
__global__ void k_gemm(const h16* __restrict__ A, const h16* __restrict__ W,
                       const float* __restrict__ bias, float* __restrict__ C,
                       int M, int N, int K, int relu, float scale) {
  int lane = threadIdx.x;
  int nBase = blockIdx.x * 16, mBase = blockIdx.y * 16;
  v8f acc = {};
  for (int k0 = 0; k0 < K; k0 += 32) {
    v16h a = frag_load(A, mBase, M, K, k0, lane);
    v16h b = frag_load(W, nBase, N, K, k0, lane);
    acc = __builtin_amdgcn_wmma_f32_16x16x32_f16(false, a, false, b, (short)0, acc,
                                                 false, false);
  }
  int col = nBase + (lane & 15);
  if (col >= N) return;
  float bv = bias ? bias[col] : 0.f;
  int row0 = mBase + ((lane >> 4) << 3);
#pragma unroll
  for (int r = 0; r < 8; ++r) {
    int row = row0 + r;
    if (row < M) {
      float v = acc[r] + bv;
      if (relu) v = v > 0.f ? v : 0.f;
      C[(size_t)row * N + col] = v * scale;
    }
  }
}

// per-graph E @ E^T / sqrt(HID) with -10 diagonal (E E^T is symmetric already)
// one wave per graph row-tile: 16x64 strip of the 64x64 output.
__global__ void k_edge_logits(const h16* __restrict__ ne16, float* __restrict__ out) {
  int lane = threadIdx.x;
  int b = blockIdx.z;
  int mBase = blockIdx.y * 16;
  const h16* E = ne16 + (size_t)b * MXN * HD;
  v8f acc0 = {}, acc1 = {}, acc2 = {}, acc3 = {};
  for (int k0 = 0; k0 < HD; k0 += 32) {
    v16h a  = frag_load_nc(E, mBase, HD, k0, lane);
    v16h b0 = frag_load_nc(E,  0, HD, k0, lane);
    v16h b1 = frag_load_nc(E, 16, HD, k0, lane);
    v16h b2 = frag_load_nc(E, 32, HD, k0, lane);
    v16h b3 = frag_load_nc(E, 48, HD, k0, lane);
    acc0 = __builtin_amdgcn_wmma_f32_16x16x32_f16(false, a, false, b0, (short)0, acc0, false, false);
    acc1 = __builtin_amdgcn_wmma_f32_16x16x32_f16(false, a, false, b1, (short)0, acc1, false, false);
    acc2 = __builtin_amdgcn_wmma_f32_16x16x32_f16(false, a, false, b2, (short)0, acc2, false, false);
    acc3 = __builtin_amdgcn_wmma_f32_16x16x32_f16(false, a, false, b3, (short)0, acc3, false, false);
  }
  int row0 = mBase + ((lane >> 4) << 3);
  float* O = out + (size_t)b * MXN * MXN;
  v8f* accs[4] = {&acc0, &acc1, &acc2, &acc3};
#pragma unroll
  for (int s = 0; s < 4; ++s) {
    int col = s * 16 + (lane & 15);
    v8f& ac = *accs[s];
#pragma unroll
    for (int r = 0; r < 8; ++r) {
      int row = row0 + r;
      O[row * MXN + col] = (row == col) ? -10.f : ac[r] * INV_SQRT_HID;
    }
  }
}

// ---- GCN ----
__global__ void k_deg(const int* __restrict__ col, float* __restrict__ deg) {
  int e = blockIdx.x * blockDim.x + threadIdx.x;
  if (e < EG) atomicAdd(&deg[col[e]], 1.f);
}
__global__ void k_rsqrt_ip(float* d, int n) {
  int i = blockIdx.x * blockDim.x + threadIdx.x;
  if (i < n) d[i] = rsqrtf(d[i]);
}
__global__ void k_gcn_edge(const int* __restrict__ row, const int* __restrict__ col,
                           const float* __restrict__ hw, const float* __restrict__ dinv,
                           float* __restrict__ out) {
  size_t idx = (size_t)blockIdx.x * blockDim.x + threadIdx.x;
  if (idx >= (size_t)EG * HD) return;
  int e = (int)(idx >> 7), ch = (int)(idx & 127);
  int r = row[e], c = col[e];
  atomicAdd(&out[(size_t)c * HD + ch], hw[(size_t)r * HD + ch] * dinv[r] * dinv[c]);
}
__global__ void k_gcn_self(const float* __restrict__ hw, const float* __restrict__ dinv,
                           const float* __restrict__ b, float* __restrict__ out) {
  int idx = blockIdx.x * blockDim.x + threadIdx.x;
  if (idx >= ND * HD) return;
  int i = idx >> 7, ch = idx & 127;
  out[idx] += hw[idx] * dinv[i] * dinv[i] + b[ch];
}

// ---- GraphNorm ----
__global__ void k_segsum(const float* __restrict__ x, const int* __restrict__ batch,
                         float* __restrict__ gsum, float* __restrict__ gcnt) {
  int idx = blockIdx.x * blockDim.x + threadIdx.x;
  if (idx >= ND * HD) return;
  int i = idx >> 7, ch = idx & 127;
  int g = batch[i];
  atomicAdd(&gsum[(size_t)g * HD + ch], x[idx]);
  if (ch == 0) atomicAdd(&gcnt[g], 1.f);
}
__global__ void k_center(const float* __restrict__ x, const int* __restrict__ batch,
                         const float* __restrict__ gmean, const float* __restrict__ gcnt,
                         const float* __restrict__ a, float* __restrict__ y) {
  int idx = blockIdx.x * blockDim.x + threadIdx.x;
  if (idx >= ND * HD) return;
  int i = idx >> 7, ch = idx & 127;
  int g = batch[i];
  float cnt = fmaxf(gcnt[g], 1.f);
  y[idx] = x[idx] - a[ch] * gmean[(size_t)g * HD + ch] / cnt;
}
__global__ void k_sumsq(const float* __restrict__ y, const int* __restrict__ batch,
                        float* __restrict__ gvar) {
  int idx = blockIdx.x * blockDim.x + threadIdx.x;
  if (idx >= ND * HD) return;
  int i = idx >> 7, ch = idx & 127;
  float v = y[idx];
  atomicAdd(&gvar[(size_t)batch[i] * HD + ch], v * v);
}
__global__ void k_gnupd(const float* __restrict__ y, const int* __restrict__ batch,
                        const float* __restrict__ gvar, const float* __restrict__ gcnt,
                        const float* __restrict__ w, const float* __restrict__ b,
                        float* __restrict__ h) {
  int idx = blockIdx.x * blockDim.x + threadIdx.x;
  if (idx >= ND * HD) return;
  int i = idx >> 7, ch = idx & 127;
  int g = batch[i];
  float cnt = fmaxf(gcnt[g], 1.f);
  float var = gvar[(size_t)g * HD + ch] / cnt;
  float t = w[ch] * y[idx] * rsqrtf(var + 1e-5f) + b[ch];
  h[idx] += t > 0.f ? t : 0.f;
}

// ---- GATv2 ----
__global__ void k_gat_logits(const int* __restrict__ row, const int* __restrict__ col,
                             const float* __restrict__ xl, const float* __restrict__ xr,
                             const float* __restrict__ att, float* __restrict__ lg) {
  int e = blockIdx.x;               // E+N blocks, 128 threads
  int t = threadIdx.x;
  int hh = t >> 5, d = t & 31;
  int r, c;
  if (e < EG) { r = row[e]; c = col[e]; } else { r = c = e - EG; }
  int ch = hh * 32 + d;
  float v = xl[(size_t)r * HD + ch] + xr[(size_t)c * HD + ch];
  v = v > 0.f ? v : 0.2f * v;       // leaky_relu 0.2
  v *= att[ch];
  for (int off = 16; off; off >>= 1) v += __shfl_down(v, off, 32);
  if (d == 0) lg[(size_t)e * 4 + hh] = v;
}
__global__ void k_segmax(const int* __restrict__ col, const float* __restrict__ lg,
                         float* __restrict__ m) {
  int idx = blockIdx.x * blockDim.x + threadIdx.x;
  if (idx >= (EG + ND) * 4) return;
  int e = idx >> 2, hh = idx & 3;
  int c = e < EG ? col[e] : e - EG;
  atomicMaxF(&m[(size_t)c * 4 + hh], lg[idx]);
}
__global__ void k_segexp(const int* __restrict__ col, const float* __restrict__ lg,
                         const float* __restrict__ m, float* __restrict__ s) {
  int idx = blockIdx.x * blockDim.x + threadIdx.x;
  if (idx >= (EG + ND) * 4) return;
  int e = idx >> 2, hh = idx & 3;
  int c = e < EG ? col[e] : e - EG;
  atomicAdd(&s[(size_t)c * 4 + hh], __expf(lg[idx] - m[(size_t)c * 4 + hh]));
}
__global__ void k_gat_agg(const int* __restrict__ row, const int* __restrict__ col,
                          const float* __restrict__ xl, const float* __restrict__ lg,
                          const float* __restrict__ m, const float* __restrict__ s,
                          float* __restrict__ out) {
  size_t idx = (size_t)blockIdx.x * blockDim.x + threadIdx.x;
  if (idx >= (size_t)(EG + ND) * HD) return;
  int e = (int)(idx >> 7), ch = (int)(idx & 127);
  int hh = ch >> 5;
  int r, c;
  if (e < EG) { r = row[e]; c = col[e]; } else { r = c = e - EG; }
  float alpha = __expf(lg[(size_t)e * 4 + hh] - m[(size_t)c * 4 + hh]) / s[(size_t)c * 4 + hh];
  atomicAdd(&out[(size_t)c * HD + ch], xl[(size_t)r * HD + ch] * alpha);
}
__global__ void k_addbias(float* __restrict__ x, const float* __restrict__ b, int n) {
  int idx = blockIdx.x * blockDim.x + threadIdx.x;
  if (idx < n) x[idx] += b[idx & 127];
}

// ---- GIN / pool / misc ----
__global__ void k_gin_agg(const int* __restrict__ row, const int* __restrict__ col,
                          const float* __restrict__ h, float* __restrict__ z) {
  size_t idx = (size_t)blockIdx.x * blockDim.x + threadIdx.x;
  if (idx >= (size_t)EG * HD) return;
  int e = (int)(idx >> 7), ch = (int)(idx & 127);
  atomicAdd(&z[(size_t)col[e] * HD + ch], h[(size_t)row[e] * HD + ch]);
}
__global__ void k_pool(const float* __restrict__ h, const int* __restrict__ batch,
                       float* __restrict__ gh) {
  int idx = blockIdx.x * blockDim.x + threadIdx.x;
  if (idx >= ND * HD) return;
  int i = idx >> 7, ch = idx & 127;
  atomicAdd(&gh[(size_t)batch[i] * HD + ch], h[idx]);
}
__global__ void k_clip(float* x, int n) {
  int i = blockIdx.x * blockDim.x + threadIdx.x;
  if (i < n) x[i] = fminf(10.f, fmaxf(-10.f, x[i]));
}

// ================= host =================
static inline unsigned nb(long long n) { return (unsigned)((n + 255) / 256); }

extern "C" void kernel_launch(void* const* d_in, const int* in_sizes, int n_in,
                              void* d_out, int out_size, void* d_ws, size_t ws_size,
                              hipStream_t stream) {
  const float* x    = (const float*)d_in[0];
  const int* ei     = (const int*)d_in[1];
  const int* row    = ei;
  const int* col    = ei + EG;
  const int* batch  = (const int*)d_in[2];
  auto P = [&](int i) { return (const float*)d_in[i]; };

  float* ws = (float*)d_ws;
  float* H   = ws + O_H;
  float* T1  = ws + O_T1;
  float* T2  = ws + O_T2;
  float* T3  = ws + O_T3;
  float* LG  = ws + O_LG;
  float* M   = ws + O_M;
  float* S   = ws + O_S;
  float* DINV= ws + O_DEG;
  float* GM  = ws + O_GM;
  float* GV  = ws + O_GV;
  float* GC  = ws + O_GC;
  float* GH  = ws + O_GH;
  float* U1  = ws + O_U1;
  float* U2  = ws + O_U2;
  float* NEB = ws + O_NEB;
  h16* A16   = (h16*)(ws + O_F16A);
  h16* W16   = (h16*)(ws + O_F16W);

  float* out = (float*)d_out;

  // generic linear: C = scale*act(A @ W^T + b), with f16 staging + WMMA
  auto lin = [&](const float* A32, const float* W32, const float* b, float* C,
                 int Mm, int Nn, int Kk, int relu, float scale) {
    k_f2h<<<nb((long long)Mm * Kk), 256, 0, stream>>>(A32, A16, Mm * Kk);
    k_f2h<<<nb((long long)Nn * Kk), 256, 0, stream>>>(W32, W16, Nn * Kk);
    if ((Mm % 16 == 0) && (Nn % 64 == 0)) {
      dim3 g(Nn / 64, Mm / 16);
      k_gemm_t64<<<g, 32, 0, stream>>>(A16, W16, b, C, Mm, Nn, Kk, relu, scale);
    } else {
      dim3 g((Nn + 15) / 16, (Mm + 15) / 16);
      k_gemm<<<g, 32, 0, stream>>>(A16, W16, b, C, Mm, Nn, Kk, relu, scale);
    }
  };

  // GraphNorm block: H += relu(GN(X)); Y is scratch for centered values
  auto gnorm = [&](const float* X, float* Y, int gi) {
    const float* w = P(9 + (gi - 1) * 3);
    const float* b = P(10 + (gi - 1) * 3);
    const float* a = P(11 + (gi - 1) * 3);
    hipMemsetAsync(GM, 0, sizeof(float) * NG * HD, stream);
    hipMemsetAsync(GC, 0, sizeof(float) * NG, stream);
    k_segsum<<<nb(ND * HD), 256, 0, stream>>>(X, batch, GM, GC);
    k_center<<<nb(ND * HD), 256, 0, stream>>>(X, batch, GM, GC, a, Y);
    hipMemsetAsync(GV, 0, sizeof(float) * NG * HD, stream);
    k_sumsq<<<nb(ND * HD), 256, 0, stream>>>(Y, batch, GV);
    k_gnupd<<<nb(ND * HD), 256, 0, stream>>>(Y, batch, GV, GC, w, b, H);
  };

  // ---- input linear (K=7, scalar) ----
  k_lin_in<<<nb(ND * HD), 256, 0, stream>>>(x, P(3), P(4), H);

  // ---- symmetric-norm degree (shared by both GCN layers) ----
  k_fill<<<nb(ND), 256, 0, stream>>>(DINV, 1.f, ND);       // self-loop
  k_deg<<<nb(EG), 256, 0, stream>>>(col, DINV);
  k_rsqrt_ip<<<nb(ND), 256, 0, stream>>>(DINV, ND);

  // ---- GCN blocks 1,2 ----
  for (int l = 0; l < 2; ++l) {
    lin(H, P(5 + 2 * l), nullptr, T1, ND, HD, HD, 0, 1.f);  // hw
    hipMemsetAsync(T2, 0, sizeof(float) * NH, stream);
    k_gcn_edge<<<nb((long long)EG * HD), 256, 0, stream>>>(row, col, T1, DINV, T2);
    k_gcn_self<<<nb(ND * HD), 256, 0, stream>>>(T1, DINV, P(6 + 2 * l), T2);
    gnorm(T2, T3, 1 + l);
  }

  // ---- GATv2 blocks 3,4 ----
  for (int l = 0; l < 2; ++l) {
    int base = 27 + 6 * l;
    lin(H, P(base + 0), P(base + 1), T1, ND, HD, HD, 0, 1.f);   // xl
    lin(H, P(base + 2), P(base + 3), T2, ND, HD, HD, 0, 1.f);   // xr
    k_gat_logits<<<EG + ND, 128, 0, stream>>>(row, col, T1, T2, P(base + 4), LG);
    k_fill<<<nb(ND * 4), 256, 0, stream>>>(M, -1e30f, ND * 4);
    hipMemsetAsync(S, 0, sizeof(float) * ND * 4, stream);
    k_segmax<<<nb((long long)(EG + ND) * 4), 256, 0, stream>>>(col, LG, M);
    k_segexp<<<nb((long long)(EG + ND) * 4), 256, 0, stream>>>(col, LG, M, S);
    hipMemsetAsync(T3, 0, sizeof(float) * NH, stream);
    k_gat_agg<<<nb((long long)(EG + ND) * HD), 256, 0, stream>>>(row, col, T1, LG, M, S, T3);
    k_addbias<<<nb(ND * HD), 256, 0, stream>>>(T3, P(base + 5), ND * HD);
    gnorm(T3, T2, 3 + l);
  }

  // ---- GIN blocks 5,6 ----
  for (int l = 0; l < 2; ++l) {
    int base = 39 + 4 * l;
    hipMemcpyAsync(T2, H, sizeof(float) * NH, hipMemcpyDeviceToDevice, stream);
    k_gin_agg<<<nb((long long)EG * HD), 256, 0, stream>>>(row, col, H, T2);
    lin(T2, P(base + 0), P(base + 1), T1, ND, 2 * HD, HD, 1, 1.f);
    lin(T1, P(base + 2), P(base + 3), T2, ND, HD, 2 * HD, 0, 1.f);
    gnorm(T2, T3, 5 + l);
  }

  // ---- global add pool ----
  hipMemsetAsync(GH, 0, sizeof(float) * NG * HD, stream);
  k_pool<<<nb(ND * HD), 256, 0, stream>>>(H, batch, GH);

  // ---- mu / logvar heads ----
  auto head = [&](int base, float* dst) {
    lin(GH, P(base + 0), P(base + 1), U1, NG, 2 * HD, HD, 1, 1.f);
    lin(U1, P(base + 2), P(base + 3), U2, NG, HD, 2 * HD, 1, 1.f);
    lin(U2, P(base + 4), P(base + 5), dst, NG, LT, HD, 0, 1.f);
  };
  head(47, out + OUT_MU);                   // mu (also z)
  head(53, out + OUT_LV);
  k_clip<<<nb(NG * LT), 256, 0, stream>>>(out + OUT_LV, NG * LT);

  // ---- decoder ----
  lin(out + OUT_MU, P(59), P(60), U1, NG, 2 * HD, LT,     1, BNF);
  lin(U1,           P(61), P(62), U2, NG, 4 * HD, 2 * HD, 1, BNF);
  lin(U2,           P(63), P(64), U1, NG, 4 * HD, 4 * HD, 1, 1.f);   // t = dec3 out
  lin(U1,           P(65), P(66), U2, NG, 4 * HD, 4 * HD, 1, 1.f);   // ne1
  lin(U2,           P(67), P(68), NEB, NG, MXN * HD, 4 * HD, 0, 1.f);// ne2 -> node_embeds
  lin(U1,           P(71), P(72), U2, NG, 2 * HD, 4 * HD, 1, 1.f);   // nc1
  lin(U2,           P(73), P(74), out + OUT_NC, NG, MXN, 2 * HD, 0, 1.f); // nc2

  // ---- node logits + edge logits from node_embeds ----
  k_f2h<<<nb((long long)NG * MXN * HD), 256, 0, stream>>>(NEB, A16, NG * MXN * HD);
  k_f2h<<<nb(FEAT * HD), 256, 0, stream>>>(P(69), W16, FEAT * HD);
  {
    dim3 g(1, (NG * MXN) / 16);
    k_gemm<<<g, 32, 0, stream>>>(A16, W16, P(70), out + OUT_NL,
                                 NG * MXN, FEAT, HD, 0, 1.f);
  }
  {
    dim3 g(1, MXN / 16, NG);
    k_edge_logits<<<g, 32, 0, stream>>>(A16, out + OUT_EL);
  }
}
